// Attn_46729244181182
// MI455X (gfx1250) — matrix operand order repaired and checked by reading, not compiled, not run on hardware
//
#include <hip/hip_runtime.h>

typedef __bf16 bf16_t;
typedef __attribute__((ext_vector_type(16))) __bf16 v16bf;
typedef __attribute__((ext_vector_type(8)))  __bf16 v8bf;
typedef __attribute__((ext_vector_type(8)))  float  v8f;

static constexpr int Bb = 32;     // batch
static constexpr int Cc = 512;    // channels
static constexpr int Nn = 1024;   // seq len (H*W)
static constexpr int Hh = 8;      // heads
static constexpr int Dh = 64;     // head dim
static constexpr int Kl = 256;    // linformer rank

// ---------------------------------------------------------------- fragments

static __device__ __forceinline__ v16bf bf_combine(v8bf lo, v8bf hi) {
  v16bf r;
#pragma unroll
  for (int i = 0; i < 8; ++i) { r[i] = lo[i]; r[i + 8] = hi[i]; }
  return r;
}

// A operand 16x32 bf16, row-major source [.., lda]:
// lane l (l<16): row m=l, K chunks [0..7] and [16..23]; lane l+16: K [8..15], [24..31]
static __device__ __forceinline__ v16bf load_a_frag(const bf16_t* A, int lda,
                                                    int row0, int col0) {
  const int l    = threadIdx.x & 31;
  const int m    = l & 15;
  const int koff = (l >> 4) * 8;
  const bf16_t* p = A + (size_t)(row0 + m) * lda + col0 + koff;
  return bf_combine(*(const v8bf*)p, *(const v8bf*)(p + 16));
}

// A fragment out of an LDS-staged [32 x 32] bf16 tile (ld = 32)
static __device__ __forceinline__ v16bf lds_a_frag(const bf16_t* smA, int mt) {
  const int l    = threadIdx.x & 31;
  const int m    = mt + (l & 15);
  const int koff = (l >> 4) * 8;
  const bf16_t* p = smA + m * 32 + koff;
  return bf_combine(*(const v8bf*)p, *(const v8bf*)(p + 16));
}

// B operand 32x16 bf16, row-major source [.., ldb]:
// lane l (l<16): col n=l, K [0..7] and [16..23]; lane l+16: K [8..15], [24..31]
static __device__ __forceinline__ v16bf load_b_frag(const bf16_t* B, int ldb,
                                                    int row0, int col0) {
  const int l = threadIdx.x & 31;
  const int koff = (l >> 4) * 8;
  const bf16_t* p = B + (size_t)(row0 + koff) * ldb + col0 + (l & 15);
  v16bf r;
#pragma unroll
  for (int i = 0; i < 8; ++i) {
    r[i] = p[(size_t)i * ldb];
    r[i + 8] = p[(size_t)(i + 16) * ldb];
  }
  return r;
}

static __device__ __forceinline__ void store_tile_bf16(bf16_t* D, int ldd,
                                                       int row0, int col0, v8f acc) {
  const int l  = threadIdx.x & 31;
  const int n  = l & 15;
  const int mb = (l >> 4) * 8;
#pragma unroll
  for (int j = 0; j < 8; ++j)
    D[(size_t)(row0 + mb + j) * ldd + col0 + n] = (bf16_t)acc[j];
}

#define WMMA_BF16(A_, B_, C_)                                                  \
  __builtin_amdgcn_wmma_f32_16x16x32_bf16(false, (A_), false, (B_), (short)0,  \
                                          (C_), false, false)

// ---------------------------------------------------------------- conversion

__global__ __launch_bounds__(256) void cvt_bf16_kernel(
    const float* __restrict__ in, bf16_t* __restrict__ out, int n) {
  int i = blockIdx.x * 256 + threadIdx.x;
  if (i < n) out[i] = (bf16_t)in[i];
}

// in [rows, cols] fp32 -> out [cols, rows] bf16
__global__ __launch_bounds__(256) void transpose_cvt_kernel(
    const float* __restrict__ in, bf16_t* __restrict__ out, int rows, int cols) {
  int i = blockIdx.x * 256 + threadIdx.x;
  if (i < rows * cols) {
    int r = i / cols, c = i % cols;
    out[(size_t)c * rows + r] = (bf16_t)in[i];
  }
}

// ---------------------------------------------------- LayerNorm + ReLU -> y

// one block (256 threads) per token row (b,n); C = 512 -> 2 elems/thread
__global__ __launch_bounds__(256) void ln_relu_kernel(
    const float* __restrict__ x, const float* __restrict__ pos,
    const float* __restrict__ g, const float* __restrict__ bta,
    bf16_t* __restrict__ y) {
  const int row = blockIdx.x;          // b*Nn + n
  const int b   = row >> 10;
  const int n   = row & (Nn - 1);
  const int t   = threadIdx.x;

  float v0 = x[((size_t)b * Cc + t)       * Nn + n] + pos[(size_t)n * Cc + t];
  float v1 = x[((size_t)b * Cc + t + 256) * Nn + n] + pos[(size_t)n * Cc + t + 256];

  float s  = v0 + v1;
  float s2 = v0 * v0 + v1 * v1;
#pragma unroll
  for (int m = 16; m; m >>= 1) {
    s  += __shfl_xor(s, m);
    s2 += __shfl_xor(s2, m);
  }
  __shared__ float ps[8], ps2[8];
  __shared__ float smu, srs;
  if ((t & 31) == 0) { ps[t >> 5] = s; ps2[t >> 5] = s2; }
  __syncthreads();
  if (t == 0) {
    float tot = 0.f, tot2 = 0.f;
#pragma unroll
    for (int i = 0; i < 8; ++i) { tot += ps[i]; tot2 += ps2[i]; }
    float mu  = tot * (1.0f / Cc);
    float var = tot2 * (1.0f / Cc) - mu * mu;
    smu = mu;
    srs = rsqrtf(var + 1e-5f);
  }
  __syncthreads();
  const float mu = smu, rs = srs;
  float o0 = fmaxf(0.f, (v0 - mu) * rs * g[t]       + bta[t]);
  float o1 = fmaxf(0.f, (v1 - mu) * rs * g[t + 256] + bta[t + 256]);
  bf16_t* yr = y + (size_t)row * Cc;
  yr[t]       = (bf16_t)o0;
  yr[t + 256] = (bf16_t)o1;
}

// ------------------------------------------- big bf16 GEMMs, 32x32 per wave
//
// block = (32,8): 8 waves; all waves share rows [tileM, tileM+32), wave w owns
// cols [blockN + w*32, +32). Per K-step the shared 32x32 A tile is staged into
// LDS with gfx1250 async global->LDS copies (ASYNCcnt), then consumed as
// ds_load A-fragments by all 8 waves (8x cross-wave reuse of A traffic).

// D[M, Cc] bf16 = A[M, Kd] x W[Kd, Cc]
__global__ __launch_bounds__(256) void gemm_bf16_kernel(
    const bf16_t* __restrict__ A, const bf16_t* __restrict__ W,
    bf16_t* __restrict__ D, int Kd) {
  __shared__ bf16_t smA[32 * 32];
  const int w     = threadIdx.y;
  const int tileM = blockIdx.y * 32;
  const int tileN = (blockIdx.x * 8 + w) * 32;

  const int tid  = w * 32 + threadIdx.x;   // 0..255
  const int arow = tid >> 3;               // 0..31
  const int acol = (tid & 7) * 4;          // 0..28 (elements; 8B per lane)
  const unsigned int ldsOff =
      (unsigned int)(uintptr_t)smA + (unsigned int)(arow * 32 + acol) * 2u;

  v8f acc00 = {}, acc01 = {}, acc10 = {}, acc11 = {};
  for (int k0 = 0; k0 < Kd; k0 += 32) {
    // async-stage A tile [32 x 32] into LDS (one b64 per lane)
    unsigned long long ga =
        (unsigned long long)(uintptr_t)(A + (size_t)(tileM + arow) * Kd + k0 + acol);
    asm volatile("global_load_async_to_lds_b64 %0, %1, off"
                 :: "v"(ldsOff), "v"(ga) : "memory");
    if (k0 + 32 < Kd)  // prefetch next weight K-block
      __builtin_prefetch(W + (size_t)(k0 + 32) * Cc + tileN, 0, 1);
    asm volatile("s_wait_asynccnt 0" ::: "memory");
    __syncthreads();

    v16bf a0 = lds_a_frag(smA, 0);
    v16bf a1 = lds_a_frag(smA, 16);
    v16bf b0 = load_b_frag(W, Cc, k0, tileN);
    v16bf b1 = load_b_frag(W, Cc, k0, tileN + 16);
    acc00 = WMMA_BF16(a0, b0, acc00);
    acc01 = WMMA_BF16(a0, b1, acc01);
    acc10 = WMMA_BF16(a1, b0, acc10);
    acc11 = WMMA_BF16(a1, b1, acc11);
    __syncthreads();  // protect smA before next iteration's async overwrite
  }
  store_tile_bf16(D, Cc, tileM,      tileN,      acc00);
  store_tile_bf16(D, Cc, tileM,      tileN + 16, acc01);
  store_tile_bf16(D, Cc, tileM + 16, tileN,      acc10);
  store_tile_bf16(D, Cc, tileM + 16, tileN + 16, acc11);
}

// ------------------------------------------------ batched projection GEMMs

// per batch b: D = projT[Kl, Nn] x S[b][Nn, Cc]   ([Kl, Cc] result)
// transposeOut=1: store as kT [B, Cc, Kl]; else v_ [B, Kl, Cc]
__global__ __launch_bounds__(256) void gemm_proj_kernel(
    const bf16_t* __restrict__ projT, const bf16_t* __restrict__ S,
    bf16_t* __restrict__ D, int transposeOut) {
  const int b     = blockIdx.z;
  const int tileN = (blockIdx.x * 8 + threadIdx.y) * 16;  // over Cc
  const int tileM = blockIdx.y * 16;                      // over Kl
  const bf16_t* Sb = S + (size_t)b * Nn * Cc;
  v8f acc = {};
  for (int k0 = 0; k0 < Nn; k0 += 32) {
    v16bf a  = load_a_frag(projT, Nn, tileM, k0);
    v16bf bf = load_b_frag(Sb, Cc, k0, tileN);
    acc = WMMA_BF16(a, bf, acc);
  }
  const int l  = threadIdx.x & 31;
  const int n  = l & 15;
  const int mb = (l >> 4) * 8;
  if (transposeOut) {
    bf16_t* p = D + ((size_t)b * Cc + tileN + n) * Kl + tileM + mb;
#pragma unroll
    for (int j = 0; j < 8; ++j) p[j] = (bf16_t)acc[j];
  } else {
#pragma unroll
    for (int j = 0; j < 8; ++j)
      D[((size_t)b * Kl + tileM + mb + j) * Cc + tileN + n] = (bf16_t)acc[j];
  }
}

// ------------------------------------------------------- fused attention

// grid (Nn/16, Hh, Bb), one wave per block.
// dots [16,256] in 16 f32 accumulators -> in-register softmax -> LDS bf16
// -> attn x V -> attnout bf16 [B, Nn, Cc]
__global__ __launch_bounds__(32) void attn_kernel(
    const bf16_t* __restrict__ q, const bf16_t* __restrict__ kT,
    const bf16_t* __restrict__ v_, bf16_t* __restrict__ attnout) {
  __shared__ bf16_t smem[16 * 256];
  const int n0 = blockIdx.x * 16;
  const int h  = blockIdx.y;
  const int b  = blockIdx.z;
  const int l  = threadIdx.x;

  // ---- dots = Q[16,64] x K^T[64,256]
  const bf16_t* qb = q + (size_t)b * Nn * Cc;
  v16bf aF0 = load_a_frag(qb, Cc, n0, h * Dh + 0);
  v16bf aF1 = load_a_frag(qb, Cc, n0, h * Dh + 32);
  const bf16_t* kTb = kT + ((size_t)b * Cc + h * Dh) * Kl;  // rows = d, ld = Kl

  v8f acc[16];
#pragma unroll
  for (int kt = 0; kt < 16; ++kt) {
    v8f a = {};
    a = WMMA_BF16(aF0, load_b_frag(kTb, Kl, 0,  kt * 16), a);
    a = WMMA_BF16(aF1, load_b_frag(kTb, Kl, 32, kt * 16), a);
    acc[kt] = a;
  }

  // ---- softmax over 256 columns (per-row: 16 tiles in-lane + 16-lane half)
  const float scale = 0.125f;  // Dh^-0.5
#pragma unroll
  for (int j = 0; j < 8; ++j) {
    float m = -3.0e38f;
#pragma unroll
    for (int kt = 0; kt < 16; ++kt) m = fmaxf(m, acc[kt][j]);
    m *= scale;
#pragma unroll
    for (int msk = 8; msk; msk >>= 1) m = fmaxf(m, __shfl_xor(m, msk));
    float s = 0.f;
#pragma unroll
    for (int kt = 0; kt < 16; ++kt) {
      float e = __expf(acc[kt][j] * scale - m);
      acc[kt][j] = e;
      s += e;
    }
#pragma unroll
    for (int msk = 8; msk; msk >>= 1) s += __shfl_xor(s, msk);
    const float inv = 1.0f / s;
    const int   row = j + 8 * (l >> 4);
#pragma unroll
    for (int kt = 0; kt < 16; ++kt)
      smem[row * 256 + kt * 16 + (l & 15)] = (bf16_t)(acc[kt][j] * inv);
  }
  __syncthreads();

  // ---- out16 = attn[16,256] x V[256,64]
  const bf16_t* vb = v_ + (size_t)b * Kl * Cc + h * Dh;  // rows = k, ld = Cc
  v8f acc2[4];
#pragma unroll
  for (int ct = 0; ct < 4; ++ct) acc2[ct] = (v8f){};
#pragma unroll
  for (int ks = 0; ks < 8; ++ks) {
    const int m    = l & 15;
    const int koff = (l >> 4) * 8;
    const bf16_t* p = &smem[m * 256 + ks * 32 + koff];
    v16bf aT = bf_combine(*(const v8bf*)p, *(const v8bf*)(p + 16));
#pragma unroll
    for (int ct = 0; ct < 4; ++ct) {
      v16bf bF = load_b_frag(vb, Cc, ks * 32, ct * 16);
      acc2[ct] = WMMA_BF16(aT, bF, acc2[ct]);
    }
  }
  const int n  = l & 15;
  const int mb = (l >> 4) * 8;
#pragma unroll
  for (int ct = 0; ct < 4; ++ct)
#pragma unroll
    for (int j = 0; j < 8; ++j)
      attnout[((size_t)b * Nn + n0 + mb + j) * Cc + h * Dh + ct * 16 + n] =
          (bf16_t)acc2[ct][j];
}

// ------------------------------------- final GEMM + residual + transpose out

// out[b,c,n] = (attnout[b*Nn+n, :] @ Wo)[c] + bo[c] + x[b,c,n] + pos[n,c]
// Same 32x32-per-wave async-staged structure as gemm_bf16_kernel.
__global__ __launch_bounds__(256) void gemm_out_kernel(
    const bf16_t* __restrict__ A, const bf16_t* __restrict__ Wo,
    const float* __restrict__ bo, const float* __restrict__ x,
    const float* __restrict__ pos, float* __restrict__ out) {
  __shared__ bf16_t smA[32 * 32];
  const int w     = threadIdx.y;
  const int tileM = blockIdx.y * 32;                      // b*Nn + n
  const int tileN = (blockIdx.x * 8 + w) * 32;            // c

  const int tid  = w * 32 + threadIdx.x;
  const int arow = tid >> 3;
  const int acol = (tid & 7) * 4;
  const unsigned int ldsOff =
      (unsigned int)(uintptr_t)smA + (unsigned int)(arow * 32 + acol) * 2u;

  v8f acc00 = {}, acc01 = {}, acc10 = {}, acc11 = {};
  for (int k0 = 0; k0 < Cc; k0 += 32) {
    unsigned long long ga =
        (unsigned long long)(uintptr_t)(A + (size_t)(tileM + arow) * Cc + k0 + acol);
    asm volatile("global_load_async_to_lds_b64 %0, %1, off"
                 :: "v"(ldsOff), "v"(ga) : "memory");
    if (k0 + 32 < Cc)
      __builtin_prefetch(Wo + (size_t)(k0 + 32) * Cc + tileN, 0, 1);
    asm volatile("s_wait_asynccnt 0" ::: "memory");
    __syncthreads();

    v16bf a0 = lds_a_frag(smA, 0);
    v16bf a1 = lds_a_frag(smA, 16);
    v16bf b0 = load_b_frag(Wo, Cc, k0, tileN);
    v16bf b1 = load_b_frag(Wo, Cc, k0, tileN + 16);
    acc00 = WMMA_BF16(a0, b0, acc00);
    acc01 = WMMA_BF16(a0, b1, acc01);
    acc10 = WMMA_BF16(a1, b0, acc10);
    acc11 = WMMA_BF16(a1, b1, acc11);
    __syncthreads();
  }

  const int l  = threadIdx.x & 31;
  const int mb = (l >> 4) * 8;
  v8f accs[4] = {acc00, acc01, acc10, acc11};
#pragma unroll
  for (int t = 0; t < 4; ++t) {
    const int mt = (t >> 1) * 16;
    const int c  = tileN + (t & 1) * 16 + (l & 15);
    const float bc = bo[c];
#pragma unroll
    for (int j = 0; j < 8; ++j) {
      const int m  = tileM + mt + mb + j;
      const int b_ = m >> 10;
      const int n  = m & (Nn - 1);
      const float xf = x[((size_t)b_ * Cc + c) * Nn + n] + pos[(size_t)n * Cc + c];
      out[((size_t)b_ * Cc + c) * Nn + n] = accs[t][j] + bc + xf;
    }
  }
}

// -------------------------------------------------------------------- launch

extern "C" void kernel_launch(void* const* d_in, const int* in_sizes, int n_in,
                              void* d_out, int out_size, void* d_ws,
                              size_t ws_size, hipStream_t stream) {
  (void)in_sizes; (void)n_in; (void)out_size; (void)ws_size;

  const float* x      = (const float*)d_in[0];
  const float* pos    = (const float*)d_in[1];
  const float* ln_g   = (const float*)d_in[2];
  const float* ln_b   = (const float*)d_in[3];
  const float* Wq     = (const float*)d_in[4];
  const float* Wk     = (const float*)d_in[5];
  const float* Wv     = (const float*)d_in[6];
  const float* proj_k = (const float*)d_in[7];
  const float* proj_v = (const float*)d_in[8];
  const float* Wo     = (const float*)d_in[9];
  const float* bo     = (const float*)d_in[10];
  float* out          = (float*)d_out;

  size_t off = 0;
  auto take = [&](size_t bytes) -> void* {
    void* r = (char*)d_ws + off;
    off += (bytes + 255) & ~(size_t)255;
    return r;
  };
  bf16_t* y_bf   = (bf16_t*)take((size_t)Bb * Nn * Cc * 2);
  bf16_t* q_bf   = (bf16_t*)take((size_t)Bb * Nn * Cc * 2);
  bf16_t* kv0_bf = (bf16_t*)take((size_t)Bb * Nn * Cc * 2);  // k0/v0, reused
  bf16_t* kT_bf  = (bf16_t*)take((size_t)Bb * Cc * Kl * 2);  // [B, C, K]
  bf16_t* v__bf  = (bf16_t*)take((size_t)Bb * Kl * Cc * 2);  // [B, K, C]
  bf16_t* Wq_bf  = (bf16_t*)take((size_t)Cc * Cc * 2);
  bf16_t* Wk_bf  = (bf16_t*)take((size_t)Cc * Cc * 2);
  bf16_t* Wv_bf  = (bf16_t*)take((size_t)Cc * Cc * 2);
  bf16_t* Wo_bf  = (bf16_t*)take((size_t)Cc * Cc * 2);
  bf16_t* pkT_bf = (bf16_t*)take((size_t)Kl * Nn * 2);       // [K, N]
  bf16_t* pvT_bf = (bf16_t*)take((size_t)Kl * Nn * 2);
  bf16_t* attnout_bf = kv0_bf;  // reuse: kv0 dead after projections

  const dim3 blk256(256);
  const dim3 gblk(32, 8);

  // weight / projection conversion
  cvt_bf16_kernel<<<dim3((Cc * Cc + 255) / 256), blk256, 0, stream>>>(Wq, Wq_bf, Cc * Cc);
  cvt_bf16_kernel<<<dim3((Cc * Cc + 255) / 256), blk256, 0, stream>>>(Wk, Wk_bf, Cc * Cc);
  cvt_bf16_kernel<<<dim3((Cc * Cc + 255) / 256), blk256, 0, stream>>>(Wv, Wv_bf, Cc * Cc);
  cvt_bf16_kernel<<<dim3((Cc * Cc + 255) / 256), blk256, 0, stream>>>(Wo, Wo_bf, Cc * Cc);
  transpose_cvt_kernel<<<dim3((Nn * Kl + 255) / 256), blk256, 0, stream>>>(proj_k, pkT_bf, Nn, Kl);
  transpose_cvt_kernel<<<dim3((Nn * Kl + 255) / 256), blk256, 0, stream>>>(proj_v, pvT_bf, Nn, Kl);

  // LayerNorm + ReLU
  ln_relu_kernel<<<dim3(Bb * Nn), blk256, 0, stream>>>(x, pos, ln_g, ln_b, y_bf);

  // Q = y Wq ; K0 = y Wk ; project; V0 = y Wv ; project
  const dim3 ggrid(Cc / 256, (Bb * Nn) / 32);   // (2, 1024): 32 rows x 256 cols/block
  const dim3 pgrid(Cc / 128, Kl / 16, Bb);
  gemm_bf16_kernel<<<ggrid, gblk, 0, stream>>>(y_bf, Wq_bf, q_bf, Cc);
  gemm_bf16_kernel<<<ggrid, gblk, 0, stream>>>(y_bf, Wk_bf, kv0_bf, Cc);
  gemm_proj_kernel<<<pgrid, gblk, 0, stream>>>(pkT_bf, kv0_bf, kT_bf, 1);
  gemm_bf16_kernel<<<ggrid, gblk, 0, stream>>>(y_bf, Wv_bf, kv0_bf, Cc);
  gemm_proj_kernel<<<pgrid, gblk, 0, stream>>>(pvT_bf, kv0_bf, v__bf, 0);

  // fused attention (dots -> softmax -> attn x V)
  attn_kernel<<<dim3(Nn / 16, Hh, Bb), dim3(32), 0, stream>>>(q_bf, kT_bf, v__bf, attnout_bf);

  // out = attnout Wo + bo + (x + pos), transposed back to [B,C,H,W]
  gemm_out_kernel<<<ggrid, gblk, 0, stream>>>(attnout_bf, Wo_bf, bo, x, pos, out);
}
